// Conv2DLayer_68178310857037
// MI455X (gfx1250) — matrix operand order repaired and checked by reading, not compiled, not run on hardware
//
#include <hip/hip_runtime.h>

// ---------------------------------------------------------------------------
// Implicit-GEMM 3x3 conv for MI455X (gfx1250, wave32).
//   Pass 1: pack fp32 weights -> bf16 hi/lo matrices in d_ws (once per call).
//   Pass 2: D(64ch x 32pos per block) = W(64x288) x im2col(288x32) + bias
//           fp32 emulated via bf16 split: a*b ~= ah*bh + ah*bl + al*bh,
//           f32 accumulation in v_wmma_f32_16x16x32_bf16.
// A fragments: bf16 global loads (L0-resident 36KB matrices, no in-loop cvt).
// B fragments: im2col tile staged in LDS as bf16 hi/lo, ds_load_b128 pairs.
// ---------------------------------------------------------------------------

typedef __attribute__((ext_vector_type(16))) __bf16 v16bf;
typedef __attribute__((ext_vector_type(8)))  __bf16 v8bf;
typedef __attribute__((ext_vector_type(8)))  float  v8f;

#define C_IN   32
#define NK     64
#define HW     128
#define KGEMM  288          // 32 * 3 * 3
#define KBLK   32
#define NKB    9            // 288 / 32
#define POSB   32           // output positions per block
#define LDSROW 296          // padded K row stride (elements), 16B-aligned rows
#define WELEMS (NK * KGEMM) // 18432 weight elements

// ---- Pass 1: split fp32 weights into bf16 hi/lo ---------------------------
__global__ __launch_bounds__(256)
void pack_weights_kernel(const float* __restrict__ kernels,
                         __bf16* __restrict__ whi,
                         __bf16* __restrict__ wlo)
{
    int i = blockIdx.x * 256 + threadIdx.x;
    if (i < WELEMS) {
        float v  = kernels[i];
        __bf16 h = (__bf16)v;
        whi[i] = h;
        wlo[i] = (__bf16)(v - (float)h);
    }
}

// ---- Pass 2: main conv ----------------------------------------------------
__global__ __launch_bounds__(256)
void conv2d_wmma_bf16x3_kernel(const float* __restrict__ x,
                               const __bf16* __restrict__ whi,
                               const __bf16* __restrict__ wlo,
                               const float* __restrict__ bias,
                               float* __restrict__ out)
{
    __shared__ __align__(16) __bf16 bhi[POSB * LDSROW];
    __shared__ __align__(16) __bf16 blo[POSB * LDSROW];

    const int bx  = blockIdx.x;
    const int n   = bx >> 9;            // 512 blocks per image (128 rows * 4 xtiles)
    const int oy  = (bx >> 2) & 127;
    const int ox0 = (bx & 3) * POSB;

    const int tid = threadIdx.x;

    // ---- Stage im2col tile (32 positions x 288 K) into LDS as bf16 hi/lo ----
    const float* xn = x + (size_t)n * (C_IN * HW * HW);
    for (int i = 0; i < (POSB * KGEMM) / 256; ++i) {
        int flat = i * 256 + tid;
        int pos  = flat & 31;           // consecutive lanes -> consecutive ix
        int kk   = flat >> 5;           // kk = c*9 + r*3 + s
        int c    = kk / 9;
        int rs   = kk - c * 9;
        int r    = rs / 3;
        int s    = rs - r * 3;
        int iy   = oy + r - 1;
        int ix   = ox0 + pos + s - 1;
        float v = 0.0f;
        if ((unsigned)iy < (unsigned)HW && (unsigned)ix < (unsigned)HW)
            v = xn[(c * HW + iy) * HW + ix];
        __bf16 h = (__bf16)v;
        __bf16 l = (__bf16)(v - (float)h);
        bhi[pos * LDSROW + kk] = h;
        blo[pos * LDSROW + kk] = l;
    }
    __syncthreads();

    // ---- Wave -> tile mapping ---------------------------------------------
    const int lane  = tid & 31;
    const int wave  = tid >> 5;
    const int mtile = wave & 3;         // 4 tiles of 16 output channels
    const int nsub  = wave >> 2;        // 2 tiles of 16 positions
    const int khalf = lane >> 4;        // lane group selects K octet/half
    const int col   = lane & 15;

    // A (16x32 bf16) lane layout: M = col; elems 0..7 -> K = khalf*8 + e,
    //                                      elems 8..15 -> K = 16 + khalf*8 + e.
    const __bf16* whr = whi + (mtile * 16 + col) * KGEMM + khalf * 8;
    const __bf16* wlr = wlo + (mtile * 16 + col) * KGEMM + khalf * 8;
    // B (32x16 bf16) lane layout: N = col; elems 0..7 -> K = khalf*8 + e,
    //                                      elems 8..15 -> K = 16 + khalf*8 + e.
    const __bf16* bhrow = &bhi[(nsub * 16 + col) * LDSROW + khalf * 8];
    const __bf16* blrow = &blo[(nsub * 16 + col) * LDSROW + khalf * 8];

    v8f acc = {};

    #pragma unroll
    for (int kb = 0; kb < NKB; ++kb) {
        // -- A fragments: four 16B bf16 loads (L0-hot), zero in-loop VALU ----
        const __bf16* ph = whr + kb * KBLK;
        const __bf16* pl = wlr + kb * KBLK;
        v8bf ah0 = *(const v8bf*)(ph);
        v8bf ah1 = *(const v8bf*)(ph + 16);
        v8bf al0 = *(const v8bf*)(pl);
        v8bf al1 = *(const v8bf*)(pl + 16);
        v16bf ahi, alo;
        #pragma unroll
        for (int e = 0; e < 8; ++e) {
            ahi[e] = ah0[e]; ahi[8 + e] = ah1[e];
            alo[e] = al0[e]; alo[8 + e] = al1[e];
        }

        // -- B fragments: two aligned 16B LDS loads per matrix ---------------
        const __bf16* bph = bhrow + kb * KBLK;
        const __bf16* bpl = blrow + kb * KBLK;
        v8bf bh0 = *(const v8bf*)(bph);
        v8bf bh1 = *(const v8bf*)(bph + 16);
        v8bf bl0 = *(const v8bf*)(bpl);
        v8bf bl1 = *(const v8bf*)(bpl + 16);
        v16bf bh, bl;
        #pragma unroll
        for (int e = 0; e < 8; ++e) {
            bh[e] = bh0[e]; bh[8 + e] = bh1[e];
            bl[e] = bl0[e]; bl[8 + e] = bl1[e];
        }

        // -- bf16x3 split-product, f32 accumulate ----------------------------
        acc = __builtin_amdgcn_wmma_f32_16x16x32_bf16(
                  false, ahi, false, bh, (short)0, acc, false, false);
        acc = __builtin_amdgcn_wmma_f32_16x16x32_bf16(
                  false, ahi, false, bl, (short)0, acc, false, false);
        acc = __builtin_amdgcn_wmma_f32_16x16x32_bf16(
                  false, alo, false, bh, (short)0, acc, false, false);
    }

    // ---- Epilogue: bias + coalesced stores --------------------------------
    // D layout: lane -> N = col (output x), VGPR r -> M = khalf*8 + r (channel).
    const int ox = ox0 + nsub * 16 + col;
    #pragma unroll
    for (int r = 0; r < 8; ++r) {
        int ch = mtile * 16 + khalf * 8 + r;
        float v = acc[r] + bias[ch];
        out[(((size_t)n * NK + ch) * HW + oy) * HW + ox] = v;
    }
}

extern "C" void kernel_launch(void* const* d_in, const int* in_sizes, int n_in,
                              void* d_out, int out_size, void* d_ws, size_t ws_size,
                              hipStream_t stream) {
    const float* x       = (const float*)d_in[0];
    const float* kernels = (const float*)d_in[1];
    const float* bias    = (const float*)d_in[2];
    float* out           = (float*)d_out;

    // Workspace layout: whi[18432] bf16 | wlo[18432] bf16  (73,728 bytes)
    __bf16* whi = (__bf16*)d_ws;
    __bf16* wlo = whi + WELEMS;

    pack_weights_kernel<<<dim3((WELEMS + 255) / 256), dim3(256), 0, stream>>>(
        kernels, whi, wlo);

    // 32 images * 128 rows * 4 x-tiles of 32 columns
    conv2d_wmma_bf16x3_kernel<<<dim3(32 * 128 * 4), dim3(256), 0, stream>>>(
        x, whi, wlo, bias, out);
}
